// StableResonantSSM_68513318306481
// MI455X (gfx1250) — compile-verified, matrix operand order checked by reading, not run on hardware
//
#include <hip/hip_runtime.h>
#include <hip/hip_bf16.h>

typedef __attribute__((ext_vector_type(16))) _Float16 v16h;
typedef __attribute__((ext_vector_type(8)))  _Float16 v8h;
typedef __attribute__((ext_vector_type(4)))  _Float16 v4h;
typedef __attribute__((ext_vector_type(8)))  float    v8f;

#define DT_STEP   0.01f
#define DAMP_MIN_ 5e-4f

constexpr int BB  = 4;
constexpr int LL  = 4096;
constexpr int DIN = 512;
constexpr int NN  = 256;
constexpr int LC  = 256;           // scan chunk length
constexpr int NC  = LL / LC;       // 16 chunks
constexpr int MM  = BB * LL;       // 16384 rows

// ---------------------------------------------------------------------------
// WMMA helpers (V_WMMA_F32_16X16X32_F16, wave32; ISA 7.12.2 layouts)
// lane r = lane&15 owns row (A) / column (B) r; hi = lane>>4;
// v16h element h holds K = (h<8 ? h : h+8) + 8*hi within the 32-wide K tile.
// ---------------------------------------------------------------------------
__device__ __forceinline__ v8f wmma16(v16h a, v16h b, v8f c) {
  return __builtin_amdgcn_wmma_f32_16x16x32_f16(
      /*neg_a=*/false, a, /*neg_b=*/false, b,
      /*c_mod=*/(short)0, c, /*reuse_a=*/false, /*reuse_b=*/false);
}

__device__ __forceinline__ v16h frag_f16(const _Float16* __restrict__ p, int hi) {
  v8h lo  = *reinterpret_cast<const v8h*>(p + 8 * hi);
  v8h hi8 = *reinterpret_cast<const v8h*>(p + 16 + 8 * hi);
  return __builtin_shufflevector(lo, hi8, 0,1,2,3,4,5,6,7,8,9,10,11,12,13,14,15);
}

// ---------------------------------------------------------------------------
// Kernel 0: bilinear discretization. A_bar[n] and A_bar^t power table (t<=LC).
// ---------------------------------------------------------------------------
__global__ void k_params(const float* __restrict__ Ar, const float* __restrict__ Ai,
                         float* __restrict__ abar_re, float* __restrict__ abar_im,
                         float* __restrict__ apow_re, float* __restrict__ apow_im) {
  int n = threadIdx.x;
  float ar = -fabsf(Ar[n]) - DAMP_MIN_;
  float ai = Ai[n];
  float dr = 1.f - 0.005f * ar, di = -0.005f * ai;       // denom = 1 - dt/2 * A
  float inv = 1.f / (dr * dr + di * di);
  float nr = 1.f + 0.005f * ar, ni = 0.005f * ai;        // numer = 1 + dt/2 * A
  float abr = (nr * dr + ni * di) * inv;
  float abi = (ni * dr - nr * di) * inv;
  abar_re[n] = abr; abar_im[n] = abi;
  float pr = 1.f, pi = 0.f;                              // A_bar^0
  for (int t = 0; t <= LC; ++t) {
    apow_re[t * NN + n] = pr; apow_im[t * NN + n] = pi;
    float qr = pr * abr - pi * abi;
    float qi = pr * abi + pi * abr;
    pr = qr; pi = qi;
  }
}

// ---------------------------------------------------------------------------
// Kernel 1: B_bar = dt*B/denom -> f16 (re/im); C -> f16. (both 131072 elems)
// ---------------------------------------------------------------------------
__global__ void k_convert(const float* __restrict__ Ar, const float* __restrict__ Ai,
                          const float* __restrict__ Bm, const float* __restrict__ Cm,
                          _Float16* __restrict__ Bre, _Float16* __restrict__ Bim,
                          _Float16* __restrict__ Ch) {
  int idx = blockIdx.x * blockDim.x + threadIdx.x;       // 0 .. NN*DIN-1
  int n = idx / DIN;
  float ar = -fabsf(Ar[n]) - DAMP_MIN_;
  float ai = Ai[n];
  float dr = 1.f - 0.005f * ar, di = -0.005f * ai;
  float inv = 1.f / (dr * dr + di * di);
  float s = DT_STEP * Bm[idx] * inv;                     // dt*B * conj(denom)/|denom|^2
  Bre[idx] = (_Float16)(s * dr);
  Bim[idx] = (_Float16)(-s * di);
  Ch[idx]  = (_Float16)Cm[idx];                          // C is DIN*NN row-major
}

// ---------------------------------------------------------------------------
// Kernel 1b: one-shot u(fp32) -> f16. Removes all in-GEMM conversions and the
// 16x redundant fp32 re-reads of u fragments (A side becomes pure v8h loads).
// ---------------------------------------------------------------------------
__global__ void k_convert_u(const float* __restrict__ u, _Float16* __restrict__ uh) {
  size_t i4 = (size_t)blockIdx.x * blockDim.x + threadIdx.x;  // 4 elems/thread
  const float4 x = reinterpret_cast<const float4*>(u)[i4];
  v4h h;
  h[0] = (_Float16)x.x; h[1] = (_Float16)x.y;
  h[2] = (_Float16)x.z; h[3] = (_Float16)x.w;
  reinterpret_cast<v4h*>(uh)[i4] = h;
}

// ---------------------------------------------------------------------------
// Kernel 2: v = u @ B_bar^T  (two real WMMA GEMMs share the A fragment)
// one wave per 16x16 tile of [MM x NN]; 4 waves/block share A rows (WGP$ reuse)
// ---------------------------------------------------------------------------
__global__ __launch_bounds__(128) void gemm_vproj(
    const _Float16* __restrict__ uh, const _Float16* __restrict__ Bre,
    const _Float16* __restrict__ Bim,
    float* __restrict__ vre, float* __restrict__ vim) {
  int w    = threadIdx.x >> 5;
  int lane = threadIdx.x & 31;
  int mt   = blockIdx.x >> 2;                 // 0..1023 : 16-row tile of BL
  int nt   = (blockIdx.x & 3) * 4 + w;        // 0..15   : 16-col tile of N
  int hi   = lane >> 4, r = lane & 15;

  const _Float16* arow  = uh  + (size_t)(mt * 16 + r) * DIN;
  const _Float16* browR = Bre + (size_t)(nt * 16 + r) * DIN;
  const _Float16* browI = Bim + (size_t)(nt * 16 + r) * DIN;
  __builtin_prefetch(arow, 0, 3);
  __builtin_prefetch(browR, 0, 3);
  __builtin_prefetch(browI, 0, 3);

  v8f accR = {}; v8f accI = {};
#pragma unroll 4
  for (int k0 = 0; k0 < DIN; k0 += 32) {
    v16h a  = frag_f16(arow  + k0, hi);
    v16h br = frag_f16(browR + k0, hi);
    v16h bi = frag_f16(browI + k0, hi);
    accR = wmma16(a, br, accR);
    accI = wmma16(a, bi, accI);
  }
  // C/D layout: lane holds column n0+r; VGPR q holds row mt*16 + q + 8*hi
  float* oR = vre + (size_t)(mt * 16 + 8 * hi) * NN + nt * 16 + r;
  float* oI = vim + (size_t)(mt * 16 + 8 * hi) * NN + nt * 16 + r;
#pragma unroll
  for (int q = 0; q < 8; ++q) {
    oR[(size_t)q * NN] = accR[q];
    oI[(size_t)q * NN] = accI[q];
  }
}

// ---------------------------------------------------------------------------
// Kernel 3: local chunk scans (zero init), in-place over v; save chunk ends.
// thread = (b, c, n), n fastest -> coalesced along N.
// ---------------------------------------------------------------------------
__global__ void scan_local(float* __restrict__ vre, float* __restrict__ vim,
                           const float* __restrict__ abar_re,
                           const float* __restrict__ abar_im,
                           float* __restrict__ ends_re, float* __restrict__ ends_im) {
  int tid = blockIdx.x * blockDim.x + threadIdx.x;   // 0 .. BB*NC*NN-1
  int n  = tid % NN;
  int bc = tid / NN;
  int c  = bc % NC;
  int b  = bc / NC;
  float ar = abar_re[n], ai = abar_im[n];
  float xr = 0.f, xi = 0.f;
  size_t base = ((size_t)(b * LL + c * LC)) * NN + n;
  for (int t = 0; t < LC; ++t) {
    size_t idx = base + (size_t)t * NN;
    float vr = vre[idx], vi = vim[idx];
    float nxr = ar * xr - ai * xi + vr;
    float nxi = ar * xi + ai * xr + vi;
    xr = nxr; xi = nxi;
    vre[idx] = xr; vim[idx] = xi;
  }
  ends_re[tid] = xr; ends_im[tid] = xi;
}

// ---------------------------------------------------------------------------
// Kernel 4: cross-chunk carry scan. carry_in[c] = state entering chunk c.
// state[c] = ends_local[c] + A^LC * state[c-1];  A^LC by 8 squarings (LC=256).
// ---------------------------------------------------------------------------
__global__ void scan_carry(const float* __restrict__ abar_re,
                           const float* __restrict__ abar_im,
                           const float* __restrict__ ends_re,
                           const float* __restrict__ ends_im,
                           float* __restrict__ carry_re, float* __restrict__ carry_im) {
  int tid = blockIdx.x * blockDim.x + threadIdx.x;   // 0 .. BB*NN-1
  int n = tid % NN;
  int b = tid / NN;
  float pr = abar_re[n], pi = abar_im[n];
#pragma unroll
  for (int i = 0; i < 8; ++i) {                      // A^(2^8) = A^256
    float qr = pr * pr - pi * pi;
    float qi = 2.f * pr * pi;
    pr = qr; pi = qi;
  }
  float cr = 0.f, ci = 0.f;
  for (int c = 0; c < NC; ++c) {
    size_t idx = ((size_t)(b * NC + c)) * NN + n;
    carry_re[idx] = cr; carry_im[idx] = ci;
    float er = ends_re[idx], ei = ends_im[idx];
    float ncr = er + pr * cr - pi * ci;
    float nci = ei + pr * ci + pi * cr;
    cr = ncr; ci = nci;
  }
}

// ---------------------------------------------------------------------------
// Kernel 5: correction + emit Re(y) as f16 for the output GEMM.
// y[b, c*LC+t, n] = v_local + A^(t+1) * carry_in[b,c,n]; only Re needed.
// ---------------------------------------------------------------------------
__global__ void scan_fix(const float* __restrict__ vre,
                         const float* __restrict__ apow_re,
                         const float* __restrict__ apow_im,
                         const float* __restrict__ carry_re,
                         const float* __restrict__ carry_im,
                         _Float16* __restrict__ yh) {
  size_t idx = (size_t)blockIdx.x * blockDim.x + threadIdx.x;  // 0 .. BB*LL*NN-1
  int n = (int)(idx % NN);
  int l = (int)((idx / NN) % LL);
  int b = (int)(idx / ((size_t)NN * LL));
  int c = l / LC, t = l % LC;
  float pr = apow_re[(t + 1) * NN + n];
  float pi = apow_im[(t + 1) * NN + n];
  size_t cidx = ((size_t)(b * NC + c)) * NN + n;
  float cr = carry_re[cidx], ci = carry_im[cidx];
  float yr = vre[idx] + pr * cr - pi * ci;
  yh[idx] = (_Float16)yr;
}

// ---------------------------------------------------------------------------
// Kernel 6: out = Re(y) @ C^T + u * D   (f16 WMMA GEMM, fp32 epilogue)
// ---------------------------------------------------------------------------
__global__ __launch_bounds__(128) void gemm_out(
    const _Float16* __restrict__ yh, const _Float16* __restrict__ Ch,
    const float* __restrict__ u, const float* __restrict__ Dv,
    float* __restrict__ out) {
  int w    = threadIdx.x >> 5;
  int lane = threadIdx.x & 31;
  int mt   = blockIdx.x >> 3;                 // 0..1023 : 16-row tile of BL
  int dt   = (blockIdx.x & 7) * 4 + w;        // 0..31   : 16-col tile of Din
  int hi   = lane >> 4, r = lane & 15;

  const _Float16* arow = yh + (size_t)(mt * 16 + r) * NN;
  const _Float16* brow = Ch + (size_t)(dt * 16 + r) * NN;  // C[d][n]: k-contig
  __builtin_prefetch(arow, 0, 3);
  __builtin_prefetch(brow, 0, 3);

  v8f acc = {};
#pragma unroll
  for (int k0 = 0; k0 < NN; k0 += 32) {
    v16h a = frag_f16(arow + k0, hi);
    v16h b = frag_f16(brow + k0, hi);
    acc = wmma16(a, b, acc);
  }
  int d = dt * 16 + r;
  float dval = Dv[d];
#pragma unroll
  for (int q = 0; q < 8; ++q) {
    size_t m = (size_t)(mt * 16 + q + 8 * hi);
    out[m * DIN + d] = acc[q] + u[m * DIN + d] * dval;
  }
}

// ---------------------------------------------------------------------------
// Host launcher. Workspace carve-out (~60 MB), all launches on `stream`.
// ---------------------------------------------------------------------------
extern "C" void kernel_launch(void* const* d_in, const int* in_sizes, int n_in,
                              void* d_out, int out_size, void* d_ws, size_t ws_size,
                              hipStream_t stream) {
  const float* u      = (const float*)d_in[0];  // [BB, LL, DIN]
  const float* A_real = (const float*)d_in[1];  // [NN]
  const float* A_imag = (const float*)d_in[2];  // [NN]
  const float* B_mat  = (const float*)d_in[3];  // [NN, DIN]
  const float* C_mat  = (const float*)d_in[4];  // [DIN, NN]
  const float* D_vec  = (const float*)d_in[5];  // [DIN]
  float* out = (float*)d_out;                   // [BB, LL, DIN]

  char* w = (char*)d_ws;
  auto carve = [&](size_t bytes) -> void* {
    void* p = (void*)w;
    w += (bytes + 255) & ~(size_t)255;
    return p;
  };
  float*    abar_re  = (float*)carve(NN * sizeof(float));
  float*    abar_im  = (float*)carve(NN * sizeof(float));
  float*    apow_re  = (float*)carve((size_t)(LC + 1) * NN * sizeof(float));
  float*    apow_im  = (float*)carve((size_t)(LC + 1) * NN * sizeof(float));
  _Float16* Bre_h    = (_Float16*)carve((size_t)NN * DIN * sizeof(_Float16));
  _Float16* Bim_h    = (_Float16*)carve((size_t)NN * DIN * sizeof(_Float16));
  _Float16* C_h      = (_Float16*)carve((size_t)DIN * NN * sizeof(_Float16));
  _Float16* u_h      = (_Float16*)carve((size_t)MM * DIN * sizeof(_Float16));
  float*    v_re     = (float*)carve((size_t)MM * NN * sizeof(float));
  float*    v_im     = (float*)carve((size_t)MM * NN * sizeof(float));
  float*    ends_re  = (float*)carve((size_t)BB * NC * NN * sizeof(float));
  float*    ends_im  = (float*)carve((size_t)BB * NC * NN * sizeof(float));
  float*    carry_re = (float*)carve((size_t)BB * NC * NN * sizeof(float));
  float*    carry_im = (float*)carve((size_t)BB * NC * NN * sizeof(float));
  _Float16* y_h      = (_Float16*)carve((size_t)MM * NN * sizeof(_Float16));

  // 0) discretize + power table
  k_params<<<1, NN, 0, stream>>>(A_real, A_imag, abar_re, abar_im, apow_re, apow_im);
  // 1) weight conversion to f16
  k_convert<<<(NN * DIN) / 256, 256, 0, stream>>>(A_real, A_imag, B_mat, C_mat,
                                                  Bre_h, Bim_h, C_h);
  // 1b) activation conversion to f16 (4 elems/thread, vectorized)
  k_convert_u<<<(unsigned)(((size_t)MM * DIN / 4) / 256), 256, 0, stream>>>(u, u_h);
  // 2) input projection: v = u @ B_bar^T  (WMMA)
  gemm_vproj<<<(MM / 16) * 4, 128, 0, stream>>>(u_h, Bre_h, Bim_h, v_re, v_im);
  // 3) chunked parallel scan
  scan_local<<<(BB * NC * NN) / 256, 256, 0, stream>>>(v_re, v_im, abar_re, abar_im,
                                                       ends_re, ends_im);
  scan_carry<<<(BB * NN) / 256, 256, 0, stream>>>(abar_re, abar_im, ends_re, ends_im,
                                                  carry_re, carry_im);
  scan_fix<<<(unsigned)(((size_t)MM * NN) / 256), 256, 0, stream>>>(
      v_re, apow_re, apow_im, carry_re, carry_im, y_h);
  // 4) output projection + skip: out = Re(y) @ C^T + u*D  (WMMA)
  gemm_out<<<(MM / 16) * 8, 128, 0, stream>>>(y_h, C_h, u, D_vec, out);
}
